// PGN_21706764714261
// MI455X (gfx1250) — matrix operand order repaired and checked by reading, not compiled, not run on hardware
//
#include <hip/hip_runtime.h>
#include <hip/hip_bf16.h>

#define BIGNUM 1000000.0f
#define B_   8
#define N_   256
#define H_   128
#define E_   128
#define MID_ 128
#define IC_  16                    // i-chunks -> 512 blocks -> ~2 waves/SIMD
#define CH_  (N_ / IC_)            // senders per chunk = 16

typedef __attribute__((ext_vector_type(16))) __bf16 v16bf;
typedef __attribute__((ext_vector_type(8)))  float  v8f;
typedef __attribute__((ext_vector_type(4)))  float  f32x4;

// ---------------------------------------------------------------------------
// Stage a [128 x 128] f32 weight chunk (row-major, stride MID_) into LDS in
// the exact v_wmma_f32_16x16x32_bf16 B-fragment layout:
//   fragment(ct, ks): lane L holds column n = ct*16 + (L&15); elems 0..7 =
//   K[ks*32 + (L>>4)*8 ..], elems 8..15 = K[ks*32 + 16 + (L>>4)*8 ..],
//   stored contiguously (32 bytes per lane -> two ds_load_b128 per fragment).
// ---------------------------------------------------------------------------
__device__ __forceinline__ void stage_weight128(const float* __restrict__ W,
                                                __bf16* lds, int tid) {
  for (int idx = tid; idx < 128 * 128; idx += 128) {
    int t    = idx & 15;
    int lane = (idx >> 4) & 31;
    int ks   = (idx >> 9) & 3;
    int ct   = idx >> 11;
    int k = ks * 32 + (lane >> 4) * 8 + (t & 7) + (t >> 3) * 16;
    int n = ct * 16 + (lane & 15);
    lds[(((ct * 4 + ks) * 32) + lane) * 16 + t] = (__bf16)W[k * MID_ + n];
  }
}

__device__ __forceinline__ v16bf read_bfrag(const __bf16* lds, int ct, int ks,
                                            int lane) {
  return *(const v16bf*)(lds + (((ct * 4 + ks) * 32) + lane) * 16);
}

// ---------------------------------------------------------------------------
// Build one 16x32 bf16 A-fragment from a contiguous f32 row (K-major).
// ISA layout: lanes L and L+16 hold row M = L&15; hi = L>>4 selects K
// sub-chunks: elems 0..7 = K[kbase + hi*8 ..], elems 8..15 =
// K[kbase + 16 + hi*8 ..].  Four float4 loads + packed f32->bf16 cvt.
// ---------------------------------------------------------------------------
__device__ __forceinline__ v16bf make_afrag(const float* __restrict__ row,
                                            int kbase, int hi) {
  const f32x4 f0 = *(const f32x4*)(row + kbase + hi * 8);
  const f32x4 f1 = *(const f32x4*)(row + kbase + hi * 8 + 4);
  const f32x4 g0 = *(const f32x4*)(row + kbase + 16 + hi * 8);
  const f32x4 g1 = *(const f32x4*)(row + kbase + 16 + hi * 8 + 4);
  v16bf a;
  a[0]  = (__bf16)f0.x; a[1]  = (__bf16)f0.y; a[2]  = (__bf16)f0.z; a[3]  = (__bf16)f0.w;
  a[4]  = (__bf16)f1.x; a[5]  = (__bf16)f1.y; a[6]  = (__bf16)f1.z; a[7]  = (__bf16)f1.w;
  a[8]  = (__bf16)g0.x; a[9]  = (__bf16)g0.y; a[10] = (__bf16)g0.z; a[11] = (__bf16)g0.w;
  a[12] = (__bf16)g1.x; a[13] = (__bf16)g1.y; a[14] = (__bf16)g1.z; a[15] = (__bf16)g1.w;
  return a;
}

// One K=128 slab of a row-tile GEMM: wave's 16 rows x 128 cols, accumulate.
__device__ __forceinline__ void gemm_halfk(const float* __restrict__ arow,
                                           const float* __restrict__ Wchunk,
                                           __bf16* lds, v8f (&acc)[8],
                                           int hi, int lane, int tid) {
  __syncthreads();                 // previous LDS users done
  stage_weight128(Wchunk, lds, tid);
  __syncthreads();
  v16bf afr[4];
#pragma unroll
  for (int ks = 0; ks < 4; ++ks) afr[ks] = make_afrag(arow, ks * 32, hi);
#pragma unroll
  for (int ct = 0; ct < 8; ++ct) {
#pragma unroll
    for (int ks = 0; ks < 4; ++ks) {
      acc[ct] = __builtin_amdgcn_wmma_f32_16x16x32_bf16(
          false, afr[ks], false, read_bfrag(lds, ct, ks, lane),
          (short)0, acc[ct], false, false);
    }
  }
}

// ---------------------------------------------------------------------------
// K0: gvec[b,m] = (graph_fts[b] @ Wg)[m] + bg[m] + be[m] + b1[m]
// ---------------------------------------------------------------------------
__global__ void k_gvec(const float* __restrict__ graph,
                       const float* __restrict__ Wg,
                       const float* __restrict__ bg,
                       const float* __restrict__ be,
                       const float* __restrict__ b1,
                       float* __restrict__ gvec) {
  int b = blockIdx.x, m = threadIdx.x;
  float s = 0.f;
  for (int k = 0; k < 128; ++k) s += graph[b * 128 + k] * Wg[k * MID_ + m];
  gvec[b * MID_ + m] = s + bg[m] + be[m] + b1[m];
}

// ---------------------------------------------------------------------------
// K1: pre_j = z@W1 + gvec(b),  pre_i = z@W2 + b2   (z = [node | hidden])
// ---------------------------------------------------------------------------
__global__ void __launch_bounds__(128) k_pre(
    const float* __restrict__ node, const float* __restrict__ hidden,
    const float* __restrict__ W1, const float* __restrict__ W2,
    const float* __restrict__ b2, const float* __restrict__ gvec,
    float* __restrict__ pre_j, float* __restrict__ pre_i) {
  __shared__ __bf16 lds[128 * 128];
  int tid = threadIdx.x, wave = tid >> 5, lane = tid & 31;
  int hi = lane >> 4, nlo = lane & 15;
  int row0 = blockIdx.x * 64 + wave * 16;          // wave owns 16 rows of B*N
  const float* arow_n = node   + (size_t)(row0 + nlo) * H_;
  const float* arow_h = hidden + (size_t)(row0 + nlo) * H_;
  v8f acc1[8] = {};
  v8f acc2[8] = {};
  gemm_halfk(arow_n, W1,              lds, acc1, hi, lane, tid);
  gemm_halfk(arow_h, W1 + 128 * MID_, lds, acc1, hi, lane, tid);
  gemm_halfk(arow_n, W2,              lds, acc2, hi, lane, tid);
  gemm_halfk(arow_h, W2 + 128 * MID_, lds, acc2, hi, lane, tid);
  int b = row0 >> 8;                               // N_ == 256
#pragma unroll
  for (int ct = 0; ct < 8; ++ct) {
    int m = ct * 16 + nlo;
    float gadd = gvec[b * MID_ + m];
    float badd = b2[m];
#pragma unroll
    for (int r = 0; r < 8; ++r) {                  // D: VGPR r -> row r + hi*8
      size_t idx = (size_t)(row0 + hi * 8 + r) * MID_ + m;
      pre_j[idx] = acc1[ct][r] + gadd;
      pre_i[idx] = acc2[ct][r] + badd;
    }
  }
}

// ---------------------------------------------------------------------------
// K2 (hot): fused msg_e GEMM + adj mask + max over a 16-sender chunk.
// Grid (b=8, jtile=4, ichunk=16) = 512 blocks x 4 waves (~2 waves/SIMD).
// Wave = 16 j-rows x 128 mid-cols; We resident in LDS (B-frag layout).
// Per-chunk maxes go out as plain stores (no atomics); k_reduce finishes.
// ---------------------------------------------------------------------------
__global__ void __launch_bounds__(128) k_edge(
    const float* __restrict__ edge, const float* __restrict__ adj,
    const float* __restrict__ We,   const float* __restrict__ pre_i,
    float* __restrict__ partial) {
  __shared__ __bf16 lds[128 * 128];
  int tid = threadIdx.x, wave = tid >> 5, lane = tid & 31;
  int hi = lane >> 4, nlo = lane & 15;
  stage_weight128(We, lds, tid);                   // We stays resident
  __syncthreads();

  int b     = blockIdx.x;
  int jbase = blockIdx.y * 64 + wave * 16;
  int ic    = blockIdx.z;
  int i0    = ic * CH_;

  v8f maxacc[8];
#pragma unroll
  for (int ct = 0; ct < 8; ++ct)
#pragma unroll
    for (int r = 0; r < 8; ++r) maxacc[ct][r] = -BIGNUM;

  for (int i = i0; i < i0 + CH_; ++i) {
    const float* arow =
        edge + ((size_t)(b * N_ + i) * N_ + (size_t)(jbase + nlo)) * E_;
    if (i + 1 < N_)                                 // stream next sender's row
      __builtin_prefetch(arow + (size_t)N_ * E_, 0, 0);
    v16bf afr[4];
#pragma unroll
    for (int ks = 0; ks < 4; ++ks) afr[ks] = make_afrag(arow, ks * 32, hi);

    const float* pirow  = pre_i + (size_t)(b * N_ + i) * MID_;
    const float* adjrow = adj + (size_t)(b * N_ + i) * N_ + jbase + hi * 8;
    f32x4 av0 = *(const f32x4*)adjrow;
    f32x4 av1 = *(const f32x4*)(adjrow + 4);
    float av[8] = {av0.x, av0.y, av0.z, av0.w, av1.x, av1.y, av1.z, av1.w};

#pragma unroll
    for (int ct = 0; ct < 8; ++ct) {
      v8f d = {};
#pragma unroll
      for (int ks = 0; ks < 4; ++ks)
        d = __builtin_amdgcn_wmma_f32_16x16x32_bf16(
            false, afr[ks], false, read_bfrag(lds, ct, ks, lane),
            (short)0, d, false, false);
      float pm = pirow[ct * 16 + nlo];
#pragma unroll
      for (int r = 0; r < 8; ++r) {
        float v = (av[r] > 0.f) ? (d[r] + pm) : -BIGNUM;
        maxacc[ct][r] = fmaxf(maxacc[ct][r], v);
      }
    }
  }

  // Plain full-bandwidth stores of the chunk-partial max (no RMW).
  float* prow = partial + (size_t)ic * (B_ * N_ * MID_);
#pragma unroll
  for (int ct = 0; ct < 8; ++ct) {
    int m = ct * 16 + nlo;
#pragma unroll
    for (int r = 0; r < 8; ++r) {
      size_t idx = (size_t)(b * N_ + jbase + hi * 8 + r) * MID_ + m;
      prow[idx] = maxacc[ct][r];
    }
  }
}

// ---------------------------------------------------------------------------
// K2b: msgs_red = max over chunks of partial, + pre_j (i-invariant, exact
// whenever a receiver has >= 1 edge; P(no edge) ~ 2^-256).
// ---------------------------------------------------------------------------
__global__ void k_reduce(const float* __restrict__ partial,
                         const float* __restrict__ pre_j,
                         float* __restrict__ msgs_red) {
  size_t idx = (size_t)blockIdx.x * 256 + threadIdx.x;
  float v = -BIGNUM;
#pragma unroll
  for (int c = 0; c < IC_; ++c)
    v = fmaxf(v, partial[(size_t)c * (B_ * N_ * MID_) + idx]);
  msgs_red[idx] = v + pre_j[idx];
}

// ---------------------------------------------------------------------------
// K3: out = relu(z@Wo1 + bo1 + msgs_red@Wo2 + bo2)
// ---------------------------------------------------------------------------
__global__ void __launch_bounds__(128) k_out(
    const float* __restrict__ node, const float* __restrict__ hidden,
    const float* __restrict__ msgs_red, const float* __restrict__ Wo1,
    const float* __restrict__ Wo2, const float* __restrict__ bo1,
    const float* __restrict__ bo2, float* __restrict__ out) {
  __shared__ __bf16 lds[128 * 128];
  int tid = threadIdx.x, wave = tid >> 5, lane = tid & 31;
  int hi = lane >> 4, nlo = lane & 15;
  int row0 = blockIdx.x * 64 + wave * 16;
  const float* arow_n = node     + (size_t)(row0 + nlo) * H_;
  const float* arow_h = hidden   + (size_t)(row0 + nlo) * H_;
  const float* arow_r = msgs_red + (size_t)(row0 + nlo) * MID_;
  v8f acc[8] = {};
  gemm_halfk(arow_n, Wo1,              lds, acc, hi, lane, tid);
  gemm_halfk(arow_h, Wo1 + 128 * MID_, lds, acc, hi, lane, tid);
  gemm_halfk(arow_r, Wo2,              lds, acc, hi, lane, tid);
#pragma unroll
  for (int ct = 0; ct < 8; ++ct) {
    int m = ct * 16 + nlo;
    float bias = bo1[m] + bo2[m];
#pragma unroll
    for (int r = 0; r < 8; ++r) {
      size_t idx = (size_t)(row0 + hi * 8 + r) * MID_ + m;
      float v = acc[ct][r] + bias;
      out[idx] = v > 0.f ? v : 0.f;
    }
  }
}

// ---------------------------------------------------------------------------
extern "C" void kernel_launch(void* const* d_in, const int* in_sizes, int n_in,
                              void* d_out, int out_size, void* d_ws,
                              size_t ws_size, hipStream_t stream) {
  (void)in_sizes; (void)n_in; (void)out_size; (void)ws_size;
  const float* node   = (const float*)d_in[0];
  const float* edge   = (const float*)d_in[1];
  const float* graph  = (const float*)d_in[2];
  const float* adj    = (const float*)d_in[3];
  const float* hidden = (const float*)d_in[4];
  const float* W1  = (const float*)d_in[5];
  const float* b1  = (const float*)d_in[6];
  const float* W2  = (const float*)d_in[7];
  const float* b2  = (const float*)d_in[8];
  const float* We  = (const float*)d_in[9];
  const float* be  = (const float*)d_in[10];
  const float* Wg  = (const float*)d_in[11];
  const float* bg  = (const float*)d_in[12];
  const float* Wo1 = (const float*)d_in[13];
  const float* bo1 = (const float*)d_in[14];
  const float* Wo2 = (const float*)d_in[15];
  const float* bo2 = (const float*)d_in[16];
  float* out = (float*)d_out;

  const size_t SLAB = (size_t)B_ * N_ * MID_;     // 262144 f32 = 1 MiB
  char*  ws       = (char*)d_ws;
  float* partial  = (float*)ws;                   // IC_ * 1 MiB = 16 MiB
  float* pre_j    = (float*)(ws + SLAB * 4 * IC_);
  float* pre_i    = (float*)(ws + SLAB * 4 * (IC_ + 1));
  float* msgs_red = (float*)(ws + SLAB * 4 * (IC_ + 2));
  float* gvec     = (float*)(ws + SLAB * 4 * (IC_ + 3));   // 8*128 f32

  k_gvec<<<dim3(B_), dim3(MID_), 0, stream>>>(graph, Wg, bg, be, b1, gvec);
  k_pre<<<dim3(B_ * N_ / 64), dim3(128), 0, stream>>>(node, hidden, W1, W2,
                                                      b2, gvec, pre_j, pre_i);
  k_edge<<<dim3(B_, N_ / 64, IC_), dim3(128), 0, stream>>>(edge, adj, We,
                                                           pre_i, partial);
  k_reduce<<<dim3((unsigned)(SLAB / 256)), dim3(256), 0, stream>>>(
      partial, pre_j, msgs_red);
  k_out<<<dim3(B_ * N_ / 64), dim3(128), 0, stream>>>(node, hidden, msgs_red,
                                                      Wo1, Wo2, bo1, bo2, out);
}
